// MnistDetector_70755291234511
// MI455X (gfx1250) — compile-verified, matrix operand order checked by reading, not run on hardware
//
#include <hip/hip_runtime.h>
#include <hip/hip_bf16.h>
#include <math.h>

typedef __attribute__((ext_vector_type(16))) _Float16 v16h;
typedef __attribute__((ext_vector_type(8)))  _Float16 v8h;
typedef __attribute__((ext_vector_type(8)))  float    v8f;
typedef __attribute__((ext_vector_type(4)))  int      v4i;

#define GAS1 __attribute__((address_space(1)))
#define LAS3 __attribute__((address_space(3)))

#if defined(__gfx1250__) && __has_builtin(__builtin_amdgcn_global_load_async_to_lds_b128)
#define HAVE_ASYNC_LDS 1
#else
#define HAVE_ASYNC_LDS 0
#endif

static __device__ inline void wait_async_lds() {
#if HAVE_ASYNC_LDS
#if __has_builtin(__builtin_amdgcn_s_wait_asynccnt)
  __builtin_amdgcn_s_wait_asynccnt(0);
#else
  asm volatile("s_wait_asynccnt 0x0" ::: "memory");
#endif
#endif
}

// ---------------------------------------------------------------------------
// Weight pre-pack: OIHW f32 -> f16 Wt[co][k], k = tap*CIN + ci  (GEMM K-major)
// ---------------------------------------------------------------------------
__global__ __launch_bounds__(256) void pack_w_f16(
    const float* __restrict__ W, _Float16* __restrict__ Wt, int COUT, int CIN) {
  int idx = blockIdx.x * blockDim.x + threadIdx.x;
  int KK = CIN * 9;
  if (idx >= COUT * KK) return;
  int co = idx / KK, k = idx - co * KK;
  int tap = k / CIN, ci = k - tap * CIN;
  Wt[idx] = (_Float16)W[(co * CIN + ci) * 9 + tap];
}

// ---------------------------------------------------------------------------
// Layer 1: 1->32 direct conv (K=9, too small for WMMA)
// ---------------------------------------------------------------------------
__global__ __launch_bounds__(256) void conv1_direct(
    const float* __restrict__ X, const float* __restrict__ W,
    const float* __restrict__ Bias, float* __restrict__ Y, int total) {
  int idx = blockIdx.x * blockDim.x + threadIdx.x;
  if (idx >= total) return;
  const int H = 112, Wd = 112, CO = 32;
  int w = idx % Wd; int t = idx / Wd;
  int h = t % H; t /= H;
  int co = t % CO; int n = t / CO;
  float s = Bias[co];
  const float* xb = X + n * H * Wd;
  #pragma unroll
  for (int kh = 0; kh < 3; ++kh)
    #pragma unroll
    for (int kw = 0; kw < 3; ++kw) {
      int hh = h + kh - 1, ww = w + kw - 1;
      if (hh >= 0 && hh < H && ww >= 0 && ww < Wd)
        s += W[co * 9 + kh * 3 + kw] * xb[hh * Wd + ww];
    }
  Y[idx] = fmaxf(s, 0.0f);
}

// ---------------------------------------------------------------------------
// 3x3 pad-1 conv + bias + relu, implicit GEMM on v_wmma_f32_16x16x32_f16.
//   Weights pre-packed f16 (async-DMA'd to LDS); activations staged f32->f16
//   transposed so both fragments are 2x ds_load_b128 per lane.
// ---------------------------------------------------------------------------
template<int CIN, int COUT>
__global__ __launch_bounds__(256) void conv3x3_wmma(
    const float* __restrict__ X, const _Float16* __restrict__ Wt,
    const float* __restrict__ Bias, float* __restrict__ Y, int H, int Wd) {
  constexpr int MT = (COUT >= 64) ? 64 : COUT;   // 64 or 32
  constexpr int MW = MT / 16;                    // 4 or 2
  constexpr int NW = 8 / MW;                     // 2 or 4
  constexpr int NT = NW * 16;                    // 32 or 64
  constexpr int KC = (CIN * 9) / 32;             // K chunks of 32
  constexpr int KK = CIN * 9;

  __shared__ __attribute__((aligned(16))) _Float16 As[MT][32];       // [m][k]
  __shared__ __attribute__((aligned(16))) _Float16 Bsw[NT][40];      // [n][k] (+16B pad)

  const int tid  = threadIdx.x;
  const int lane = tid & 31;
  const int wid  = tid >> 5;
  const int mi   = wid % MW;
  const int ni   = wid / MW;

  const int wTiles  = (Wd + NT - 1) / NT;
  const int coTiles = (COUT + MT - 1) / MT;
  int bid = blockIdx.x;
  int wt = bid % wTiles;  bid /= wTiles;
  int h  = bid % H;       bid /= H;
  int ct = bid % coTiles; bid /= coTiles;
  int n  = bid;
  const int w0  = wt * NT;
  const int co0 = ct * MT;

  v8f acc = {};

  for (int c = 0; c < KC; ++c) {
    const int tap = (c * 32) / CIN;
    const int ci0 = (c * 32) % CIN;
    const int dh  = tap / 3 - 1;
    const int dw  = tap % 3 - 1;

    // --- Weight tile: MT rows x 64B, one b128 per lane, async DMA to LDS ---
    {
      const int m = tid >> 2, sub = tid & 3;     // 4 lanes per row
      if (m < MT) {
        const _Float16* g = Wt + (size_t)(co0 + m) * KK + c * 32 + sub * 8;
#if HAVE_ASYNC_LDS
        __builtin_amdgcn_global_load_async_to_lds_b128(
            (GAS1 v4i*)(void*)g,
            (LAS3 v4i*)(void*)&As[m][sub * 8],
            0, 0);
#else
        *(v8h*)&As[m][sub * 8] = *(const v8h*)g;
#endif
      }
    }

    // --- Activation tile, transposed [n][k], f32 -> f16, pad-1 bounds -> 0 ---
    const int hh = h + dh;
    for (int e = tid; e < 32 * NT; e += 256) {
      int kk = e / NT, j = e - kk * NT;
      int ww = w0 + j + dw;
      float v = 0.0f;
      if (hh >= 0 && hh < H && ww >= 0 && ww < Wd)
        v = X[((n * CIN + ci0 + kk) * H + hh) * Wd + ww];
      Bsw[j][kk] = (_Float16)v;
    }

    if (c + 1 < KC)
      __builtin_prefetch(Wt + (size_t)co0 * KK + (c + 1) * 32, 0, 1);

    wait_async_lds();
    __syncthreads();

    // --- Fragments: 16-bit layout = two contiguous 8-half runs per lane ---
    const int hf   = lane >> 4;
    const int mrow = (lane & 15) + mi * 16;
    const int ncol = (lane & 15) + ni * 16;
    v8h a0 = *(const v8h*)&As[mrow][8 * hf];
    v8h a1 = *(const v8h*)&As[mrow][16 + 8 * hf];
    v8h b0 = *(const v8h*)&Bsw[ncol][8 * hf];
    v8h b1 = *(const v8h*)&Bsw[ncol][16 + 8 * hf];
    v16h a = __builtin_shufflevector(a0, a1, 0, 1, 2, 3, 4, 5, 6, 7,
                                     8, 9, 10, 11, 12, 13, 14, 15);
    v16h b = __builtin_shufflevector(b0, b1, 0, 1, 2, 3, 4, 5, 6, 7,
                                     8, 9, 10, 11, 12, 13, 14, 15);
    acc = __builtin_amdgcn_wmma_f32_16x16x32_f16(
        false, a, false, b, (short)0, acc, false, false);
    __syncthreads();
  }

  // Store D (VGPR r -> M=r+8*half, N=lane&15) with bias + relu
  const int hf = lane >> 4;
  const int nc = lane & 15;
  #pragma unroll
  for (int r = 0; r < 8; ++r) {
    int m  = r + 8 * hf;
    int co = co0 + mi * 16 + m;
    int w  = w0 + ni * 16 + nc;
    if (co < COUT && w < Wd) {
      float v = acc[r] + Bias[co];
      Y[((n * COUT + co) * H + h) * Wd + w] = fmaxf(v, 0.0f);
    }
  }
}

// ---------------------------------------------------------------------------
// 2x2 max pool, stride 2
// ---------------------------------------------------------------------------
__global__ __launch_bounds__(256) void maxpool2(
    const float* __restrict__ X, float* __restrict__ Y,
    int total, int Ho, int Wo) {
  int idx = blockIdx.x * blockDim.x + threadIdx.x;
  if (idx >= total) return;
  int wo = idx % Wo; int t = idx / Wo;
  int ho = t % Ho;   int nc = t / Ho;
  int Wi = 2 * Wo;
  const float* p = X + ((size_t)nc * (2 * Ho) + 2 * ho) * Wi + 2 * wo;
  Y[idx] = fmaxf(fmaxf(p[0], p[1]), fmaxf(p[Wi], p[Wi + 1]));
}

// ---------------------------------------------------------------------------
// rw3: 256->45 1x1 conv + tanh, written into 16x16 grid, edge-replicated past 14x14
// ---------------------------------------------------------------------------
__global__ __launch_bounds__(256) void conv1x1_tanh_pad(
    const float* __restrict__ X, const float* __restrict__ W,
    const float* __restrict__ Bias, float* __restrict__ Y) {
  int idx = blockIdx.x * blockDim.x + threadIdx.x;   // B*45*256
  if (idx >= 16 * 45 * 256) return;
  int j = idx & 15, i = (idx >> 4) & 15;
  int t = idx >> 8; int ch = t % 45; int n = t / 45;
  int ii = min(i, 13), jj = min(j, 13);
  const float* xp = X + (size_t)(n * 256) * 196 + ii * 14 + jj;
  const float* wp = W + ch * 256;
  float s = Bias[ch];
  for (int c = 0; c < 256; ++c) s += wp[c] * xp[(size_t)c * 196];
  Y[idx] = tanhf(s);
}

// ---------------------------------------------------------------------------
// Heads
// ---------------------------------------------------------------------------
__device__ inline void anchor_wh(int q, float& aw, float& ah) {
  const float S[3] = {0.15f, 0.45f, 0.75f};
  const float R[3] = {0.70710678f, 1.0f, 1.41421356f};
  float s = S[q / 3], r = R[q % 3];
  aw = s * r; ah = s / r;
}

__global__ __launch_bounds__(256) void conf_sigmoid(
    const float* __restrict__ rpad, float* __restrict__ conf) {
  int idx = blockIdx.x * blockDim.x + threadIdx.x;   // B*2304
  if (idx >= 16 * 2304) return;
  int a = idx % 2304, b = idx / 2304;
  int q = a >> 8, ij = a & 255;
  float v = rpad[(b * 45 + q) * 256 + ij];
  conf[idx] = 1.0f / (1.0f + expf(-v));
}

__global__ __launch_bounds__(256) void iou_kernel(
    const float* __restrict__ gt /*B,4,8*/, float* __restrict__ iou /*B,8,2304*/) {
  int idx = blockIdx.x * blockDim.x + threadIdx.x;   // B*2304
  if (idx >= 16 * 2304) return;
  int a = idx % 2304, b = idx / 2304;
  int q = a >> 8, i = (a >> 4) & 15, j = a & 15;
  float aw, ah; anchor_wh(q, aw, ah);
  float cx = ((float)i + 0.5f) / 16.0f, cy = ((float)j + 0.5f) / 16.0f;
  float ax1 = cx - aw * 0.5f, ay1 = cy - ah * 0.5f;
  float ax2 = cx + aw * 0.5f, ay2 = cy + ah * 0.5f;
  float aa = aw * ah;
  for (int n = 0; n < 8; ++n) {
    float gx1 = gt[(b * 4 + 0) * 8 + n];
    float gy1 = gt[(b * 4 + 1) * 8 + n];
    float gx2 = gt[(b * 4 + 2) * 8 + n];
    float gy2 = gt[(b * 4 + 3) * 8 + n];
    float ix1 = fmaxf(gx1, ax1), iy1 = fmaxf(gy1, ay1);
    float ix2 = fminf(gx2, ax2), iy2 = fminf(gy2, ay2);
    float inter = fmaxf(ix2 - ix1, 0.0f) * fmaxf(iy2 - iy1, 0.0f);
    float ga = (gx2 - gx1) * (gy2 - gy1);
    iou[(size_t)(b * 8 + n) * 2304 + a] = inter / (ga + aa - inter + 1e-8f);
  }
}

__global__ __launch_bounds__(256) void iou_argmax(
    const float* __restrict__ iou, int* __restrict__ best) {
  __shared__ float sv[256];
  __shared__ int   si[256];
  int row = blockIdx.x, tid = threadIdx.x;
  const float* p = iou + (size_t)row * 2304;
  float bv = -1e30f; int bi = 0;
  for (int a = tid; a < 2304; a += 256) { float v = p[a]; if (v > bv) { bv = v; bi = a; } }
  sv[tid] = bv; si[tid] = bi; __syncthreads();
  for (int s = 128; s > 0; s >>= 1) {
    if (tid < s) {
      if (sv[tid + s] > sv[tid] || (sv[tid + s] == sv[tid] && si[tid + s] < si[tid])) {
        sv[tid] = sv[tid + s]; si[tid] = si[tid + s];
      }
    }
    __syncthreads();
  }
  if (tid == 0) best[row] = si[0];
}

__global__ void iou_adjust(float* __restrict__ iou, const int* __restrict__ best) {
  int r = blockIdx.x * blockDim.x + threadIdx.x;
  if (r < 128) {
    float* p = &iou[(size_t)r * 2304 + best[r]];
    *p = fmaxf(*p, 0.5f);
  }
}

__global__ __launch_bounds__(256) void iou_max_k(
    const float* __restrict__ iou, float* __restrict__ out) {
  int idx = blockIdx.x * blockDim.x + threadIdx.x;   // B*2304
  if (idx >= 16 * 2304) return;
  int a = idx % 2304, b = idx / 2304;
  float m = -1e30f;
  for (int n = 0; n < 8; ++n) m = fmaxf(m, iou[(size_t)(b * 8 + n) * 2304 + a]);
  out[idx] = m;
}

// ---------------------------------------------------------------------------
// Per-batch top-64 + box decode + integer crop rects
// ---------------------------------------------------------------------------
__global__ __launch_bounds__(256) void topk_decode(
    const float* __restrict__ conf, const float* __restrict__ rpad,
    float* __restrict__ boxes, int* __restrict__ pxi) {
  __shared__ float cl[2304];
  __shared__ float sv[256];
  __shared__ int   si[256];
  int b = blockIdx.x, tid = threadIdx.x;
  for (int a = tid; a < 2304; a += 256) cl[a] = conf[b * 2304 + a];
  __syncthreads();
  for (int k = 0; k < 64; ++k) {
    float bv = -1e30f; int bi = 0;
    for (int a = tid; a < 2304; a += 256) { float v = cl[a]; if (v > bv) { bv = v; bi = a; } }
    sv[tid] = bv; si[tid] = bi; __syncthreads();
    for (int s = 128; s > 0; s >>= 1) {
      if (tid < s) {
        if (sv[tid + s] > sv[tid] || (sv[tid + s] == sv[tid] && si[tid + s] < si[tid])) {
          sv[tid] = sv[tid + s]; si[tid] = si[tid + s];
        }
      }
      __syncthreads();
    }
    if (tid == 0) {
      int idx = si[0];
      cl[idx] = -1e30f;
      int q = idx >> 8, i = (idx >> 4) & 15, j = idx & 15, ij = idx & 255;
      float d0 = rpad[(b * 45 +  9 + q) * 256 + ij];
      float d1 = rpad[(b * 45 + 18 + q) * 256 + ij];
      float d2 = rpad[(b * 45 + 27 + q) * 256 + ij];
      float d3 = rpad[(b * 45 + 36 + q) * 256 + ij];
      float aw, ah; anchor_wh(q, aw, ah);
      float cx = ((float)i + 0.5f) / 16.0f + d0 * aw;
      float cy = ((float)j + 0.5f) / 16.0f + d1 * ah;
      float bw = aw * expf(d2);
      float bh = ah * expf(d3);
      float bx[4] = {cx - bw * 0.5f, cy - bh * 0.5f, cx + bw * 0.5f, cy + bh * 0.5f};
      int o = (b * 64 + k) * 4;
      int p[4];
      #pragma unroll
      for (int c = 0; c < 4; ++c) {
        boxes[o + c] = bx[c];
        int pv = (int)rintf(bx[c] * 16.0f);
        p[c] = min(15, max(0, pv));
      }
      int lx = min(p[0], 14), ly = min(p[1], 14);
      int hx = max(p[2], lx + 1), hy = max(p[3], ly + 1);
      pxi[o + 0] = lx; pxi[o + 1] = ly; pxi[o + 2] = hx; pxi[o + 3] = hy;
    }
    __syncthreads();
  }
}

// ---------------------------------------------------------------------------
// Bilinear crop-resize: one block per (batch, box); output 64 x 28 x 28
// ---------------------------------------------------------------------------
__global__ __launch_bounds__(256) void crop_resize_k(
    const float* __restrict__ feat, const int* __restrict__ pxi,
    float* __restrict__ regions) {
  __shared__ int   xi0[28], xi1[28], yi0[28], yi1[28];
  __shared__ float xt[28], yt[28];
  int blk = blockIdx.x;           // b*64 + k
  int b = blk >> 6;
  int tid = threadIdx.x;
  const int* pb = pxi + blk * 4;
  int xlo = pb[0], ylo = pb[1], xhi = pb[2], yhi = pb[3];
  if (tid < 28) {
    float n = (float)(xhi - xlo + 1);
    float g = ((float)tid + 0.5f) * n / 28.0f - 0.5f;
    float s = (float)xlo + fminf(fmaxf(g, 0.0f), n - 1.0f);
    int i0 = (int)floorf(s); i0 = min(27, max(0, i0));
    xi0[tid] = i0; xi1[tid] = min(i0 + 1, 27); xt[tid] = s - (float)i0;
  } else if (tid >= 32 && tid < 60) {
    int o = tid - 32;
    float n = (float)(yhi - ylo + 1);
    float g = ((float)o + 0.5f) * n / 28.0f - 0.5f;
    float s = (float)ylo + fminf(fmaxf(g, 0.0f), n - 1.0f);
    int i0 = (int)floorf(s); i0 = min(27, max(0, i0));
    yi0[o] = i0; yi1[o] = min(i0 + 1, 27); yt[o] = s - (float)i0;
  }
  __syncthreads();
  const float* fb = feat + (size_t)b * 64 * 784;
  float* ob = regions + (size_t)blk * 64 * 784;
  for (int e = tid; e < 64 * 784; e += 256) {
    int c = e / 784; int rem = e - c * 784;
    int ox = rem / 28; int oy = rem - ox * 28;
    const float* fc = fb + (size_t)c * 784;
    int x0 = xi0[ox], x1 = xi1[ox]; float tx = xt[ox];
    int y0 = yi0[oy], y1 = yi1[oy]; float ty = yt[oy];
    float f00 = fc[x0 * 28 + y0], f01 = fc[x0 * 28 + y1];
    float f10 = fc[x1 * 28 + y0], f11 = fc[x1 * 28 + y1];
    ob[e] = f00 * (1.0f - tx) * (1.0f - ty) + f01 * (1.0f - tx) * ty
          + f10 * tx * (1.0f - ty) + f11 * tx * ty;
  }
}

// ---------------------------------------------------------------------------
// Host-side orchestration
// ---------------------------------------------------------------------------
static inline int cdiv(int a, int b) { return (a + b - 1) / b; }

template<int CIN, int COUT>
static void launch_conv(const float* X, const float* W, _Float16* WtScratch,
                        const float* Bias, float* Y, int H, int Wd, hipStream_t s) {
  int ktotal = COUT * CIN * 9;
  pack_w_f16<<<cdiv(ktotal, 256), 256, 0, s>>>(W, WtScratch, COUT, CIN);
  constexpr int MT = (COUT >= 64) ? 64 : COUT;
  constexpr int NT = (8 / (MT / 16)) * 16;
  int wTiles  = (Wd + NT - 1) / NT;
  int coTiles = (COUT + MT - 1) / MT;
  int blocks  = 16 * H * wTiles * coTiles;
  conv3x3_wmma<CIN, COUT><<<blocks, 256, 0, s>>>(X, WtScratch, Bias, Y, H, Wd);
}

extern "C" void kernel_launch(void* const* d_in, const int* in_sizes, int n_in,
                              void* d_out, int out_size, void* d_ws, size_t ws_size,
                              hipStream_t stream) {
  (void)in_sizes; (void)n_in; (void)out_size; (void)ws_size;
  const float* x   = (const float*)d_in[0];
  const float* gt  = (const float*)d_in[1];
  const float* fw1 = (const float*)d_in[2];  const float* fb1 = (const float*)d_in[3];
  const float* fw2 = (const float*)d_in[4];  const float* fb2 = (const float*)d_in[5];
  const float* fw3 = (const float*)d_in[6];  const float* fb3 = (const float*)d_in[7];
  const float* fw4 = (const float*)d_in[8];  const float* fb4 = (const float*)d_in[9];
  const float* rw1 = (const float*)d_in[10]; const float* rb1 = (const float*)d_in[11];
  const float* rw2 = (const float*)d_in[12]; const float* rb2 = (const float*)d_in[13];
  const float* rw3 = (const float*)d_in[14]; const float* rb3 = (const float*)d_in[15];

  // workspace layout (floats)
  float* ws = (float*)d_ws;
  float* y1   = ws;                    // 16*32*112*112 = 6,422,528
  float* y2   = y1   + 6422528;
  float* p1   = y2   + 6422528;        // 1,605,632
  float* y3   = p1   + 1605632;        // 3,211,264
  float* y4   = y3   + 3211264;
  float* feat = y4   + 3211264;        //   802,816
  float* r1   = feat + 802816;         // 1,605,632
  float* r2   = r1   + 1605632;        // 3,211,264
  float* rp   = r2   + 3211264;        //   802,816
  float* rpad = rp   + 802816;         //   184,320
  float* iou  = rpad + 184320;         //   294,912
  int*   best = (int*)(iou + 294912);  // 128
  int*   pxi  = best + 128;            // 4096
  _Float16* wtp = (_Float16*)(pxi + 4096); // 294,912 halves max (16B-aligned offset)

  // output layout (floats): regions | conf | boxes | iou_max
  const size_t REG_SZ  = (size_t)16 * 64 * 64 * 28 * 28;
  const size_t CONF_SZ = (size_t)16 * 2304;
  const size_t BOX_SZ  = (size_t)16 * 64 * 4;
  float* out_regions = (float*)d_out;
  float* out_conf    = out_regions + REG_SZ;
  float* out_boxes   = out_conf + CONF_SZ;
  float* out_ioumax  = out_boxes + BOX_SZ;

  // backbone
  {
    int total = 16 * 32 * 112 * 112;
    conv1_direct<<<cdiv(total, 256), 256, 0, stream>>>(x, fw1, fb1, y1, total);
  }
  launch_conv<32, 32>(y1, fw2, wtp, fb2, y2, 112, 112, stream);
  {
    int total = 16 * 32 * 56 * 56;
    maxpool2<<<cdiv(total, 256), 256, 0, stream>>>(y2, p1, total, 56, 56);
  }
  launch_conv<32, 64>(p1, fw3, wtp, fb3, y3, 56, 56, stream);
  launch_conv<64, 64>(y3, fw4, wtp, fb4, y4, 56, 56, stream);
  {
    int total = 16 * 64 * 28 * 28;
    maxpool2<<<cdiv(total, 256), 256, 0, stream>>>(y4, feat, total, 28, 28);
  }
  launch_conv<64, 128>(feat, rw1, wtp, rb1, r1, 28, 28, stream);
  launch_conv<128, 256>(r1, rw2, wtp, rb2, r2, 28, 28, stream);
  {
    int total = 16 * 256 * 14 * 14;
    maxpool2<<<cdiv(total, 256), 256, 0, stream>>>(r2, rp, total, 14, 14);
  }
  {
    int total = 16 * 45 * 256;
    conv1x1_tanh_pad<<<cdiv(total, 256), 256, 0, stream>>>(rp, rw3, rb3, rpad);
  }

  // heads
  {
    int total = 16 * 2304;
    conf_sigmoid<<<cdiv(total, 256), 256, 0, stream>>>(rpad, out_conf);
    iou_kernel<<<cdiv(total, 256), 256, 0, stream>>>(gt, iou);
  }
  iou_argmax<<<128, 256, 0, stream>>>(iou, best);
  iou_adjust<<<1, 128, 0, stream>>>(iou, best);
  {
    int total = 16 * 2304;
    iou_max_k<<<cdiv(total, 256), 256, 0, stream>>>(iou, out_ioumax);
  }
  topk_decode<<<16, 256, 0, stream>>>(out_conf, rpad, out_boxes, pxi);
  crop_resize_k<<<16 * 64, 256, 0, stream>>>(feat, pxi, out_regions);
}
